// AttentionBlock_3504693313953
// MI455X (gfx1250) — compile-verified, matrix operand order checked by reading, not run on hardware
//
#include <hip/hip_runtime.h>
#include <hip/hip_bf16.h>

typedef __attribute__((ext_vector_type(16))) _Float16 v16h;
typedef __attribute__((ext_vector_type(8)))  _Float16 v8h;
typedef __attribute__((ext_vector_type(8)))  float    v8f;

#define LDT 40  // LDS row stride in halves (32 data + 8 pad -> 80B rows, 16B-aligned chunks)

// ---- CDNA5 async global->LDS (bypasses VGPRs, tracked by ASYNCcnt) ----
__device__ __forceinline__ void async_ld_b128(unsigned int lds_byte_addr, const _Float16* g) {
    asm volatile("global_load_async_to_lds_b128 %0, %1, off"
                 :: "v"(lds_byte_addr), "v"((unsigned long long)(uintptr_t)g)
                 : "memory");
}
__device__ __forceinline__ void wait_async() {
    asm volatile("s_wait_asynccnt 0x0" ::: "memory");
}

// Build a 16x32-f16 WMMA fragment for one matrix row from LDS.
// CDNA5 layout: lanes 0-15 hold K={0..7,16..23}, lanes 16-31 hold K={8..15,24..31}.
__device__ __forceinline__ v16h make_frag(const _Float16* lds, int row, int kSel) {
    const v8h lo = *(const v8h*)(lds + row * LDT + kSel);
    const v8h hi = *(const v8h*)(lds + row * LDT + 16 + kSel);
    v16h f;
#pragma unroll
    for (int i = 0; i < 8; ++i) { f[i] = lo[i]; f[i + 8] = hi[i]; }
    return f;
}

enum { STORE_F16 = 0, STORE_F32_RESID = 1 };

// D[m,n] = scale * sum_k A[m,k]*B[n,k]  (+ bias[n]) ; NT GEMM, f16 in, f32 accum.
// WG tile 128x128, 8 waves in 4(M)x2(N), wave tile 32x64 (2x4 WMMA tiles).
// K consumed 64 at a time through double-buffered LDS filled by async loads.
// All problem dims are multiples of 128/64 -> no bounds checks.
template <int MODE, bool HAS_BIAS>
__global__ __launch_bounds__(256)
void gemm_nt_wmma(const _Float16* __restrict__ A, const _Float16* __restrict__ B,
                  const float* __restrict__ bias,
                  _Float16* __restrict__ D16, float* __restrict__ D32,
                  const float* __restrict__ resid,
                  int K, int ldd, float scale,
                  long strideA, long strideB, long strideD)
{
    constexpr int SUB = 128 * LDT;                    // halves per 128x32 subtile
    __shared__ __align__(16) _Float16 sMem[8 * SUB];  // [A buf0 s0,s1 | A buf1 s0,s1 | B ...]

    const int tid   = threadIdx.x;
    const int lane  = tid & 31;
    const int wave  = tid >> 5;
    const int waveM = wave & 3;       // 0..3
    const int waveN = wave >> 2;      // 0..1
    const int z     = blockIdx.z;
    const long blockM = (long)blockIdx.y * 128;
    const long blockN = (long)blockIdx.x * 128;

    A += (long)z * strideA;
    B += (long)z * strideB;

    // Staging assignment: each thread owns 16 halves (two b128) of A and of B per subtile.
    const int ldRow = tid >> 1;          // 0..127
    const int ldCol = (tid & 1) * 16;    // 0 or 16

    const _Float16* gA = A + (blockM + ldRow) * (long)K + ldCol;
    const _Float16* gB = B + (blockN + ldRow) * (long)K + ldCol;

    const unsigned int ldsBase = (unsigned int)(uintptr_t)(&sMem[0]);
    const unsigned int stRow   = (unsigned int)(ldRow * LDT + ldCol) * 2u;  // bytes in subtile

    auto issue = [&](int buf, int k0) {
#pragma unroll
        for (int sub = 0; sub < 2; ++sub) {
            const int kk = k0 + sub * 32;
            const unsigned int aOff = ldsBase + (unsigned int)((buf * 2 + sub) * SUB) * 2u + stRow;
            const unsigned int bOff = ldsBase + (unsigned int)((4 + buf * 2 + sub) * SUB) * 2u + stRow;
            async_ld_b128(aOff,      gA + kk);
            async_ld_b128(aOff + 16, gA + kk + 8);
            async_ld_b128(bOff,      gB + kk);
            async_ld_b128(bOff + 16, gB + kk + 8);
        }
    };

    v8f acc[2][4] = {};
    const int kSel = (lane < 16) ? 0 : 8;
    const int lr   = lane & 15;

    issue(0, 0);                 // prologue: fill buffer 0
    int buf = 0;
    for (int k0 = 0; k0 < K; k0 += 64, buf ^= 1) {
        wait_async();            // my async writes for buf are done
        __syncthreads();         // everyone's are done; everyone finished reading buf^1
        if (k0 + 64 < K) issue(buf ^ 1, k0 + 64);   // overlap next tile with compute

#pragma unroll
        for (int sub = 0; sub < 2; ++sub) {
            const _Float16* tA = sMem + (buf * 2 + sub) * SUB;
            const _Float16* tB = sMem + (4 + buf * 2 + sub) * SUB;
            v16h af[2], bf[4];
#pragma unroll
            for (int mt = 0; mt < 2; ++mt)
                af[mt] = make_frag(tA, waveM * 32 + mt * 16 + lr, kSel);
#pragma unroll
            for (int nt = 0; nt < 4; ++nt)
                bf[nt] = make_frag(tB, waveN * 64 + nt * 16 + lr, kSel);
#pragma unroll
            for (int mt = 0; mt < 2; ++mt)
#pragma unroll
                for (int nt = 0; nt < 4; ++nt)
                    acc[mt][nt] = __builtin_amdgcn_wmma_f32_16x16x32_f16(
                        false, af[mt], false, bf[nt], (short)0, acc[mt][nt], false, false);
        }
    }

    // C/D layout: VGPR r -> (M = r or r+8 by lane half, N = lane&15)
#pragma unroll
    for (int mt = 0; mt < 2; ++mt) {
#pragma unroll
        for (int nt = 0; nt < 4; ++nt) {
#pragma unroll
            for (int r = 0; r < 8; ++r) {
                const long gRow = blockM + waveM * 32 + mt * 16 + ((lane < 16) ? r : r + 8);
                const int  gCol = (int)blockN + waveN * 64 + nt * 16 + lr;
                float v = acc[mt][nt][r] * scale;
                if (HAS_BIAS) v += bias[gCol];
                if (MODE == STORE_F16) {
                    D16[(long)z * strideD + gRow * (long)ldd + gCol] = (_Float16)v;
                } else {
                    // gRow = b*1024 + token ; output is [B, C, N] with residual
                    const long b   = gRow >> 10;
                    const long n   = gRow & 1023;
                    const long idx = (b * 512 + gCol) * 1024 + n;
                    D32[idx] = resid[idx] + v;
                }
            }
        }
    }
}

// GroupNorm(32 groups) over [B=32, C=512, N=1024]; writes token-major f16 h[B*N, C].
__global__ __launch_bounds__(256)
void groupnorm_to_f16(const float* __restrict__ x, const float* __restrict__ gscale,
                      const float* __restrict__ gbias, _Float16* __restrict__ h16)
{
    const int b = blockIdx.x >> 5;
    const int g = blockIdx.x & 31;
    const int tid = threadIdx.x;
    const long base = ((long)b * 512 + (long)g * 16) * 1024;

    float s = 0.f, sq = 0.f;
    for (int i = tid; i < 16 * 1024; i += 256) {
        const float v = x[base + i];
        s += v; sq += v * v;
    }
    __shared__ float rs[256], rq[256];
    rs[tid] = s; rq[tid] = sq;
    __syncthreads();
    for (int off = 128; off > 0; off >>= 1) {
        if (tid < off) { rs[tid] += rs[tid + off]; rq[tid] += rq[tid + off]; }
        __syncthreads();
    }
    const float mean = rs[0] * (1.f / 16384.f);
    const float var  = rq[0] * (1.f / 16384.f) - mean * mean;
    const float rstd = rsqrtf(var + 1e-5f);

    for (int i = tid; i < 16 * 1024; i += 256) {
        const int cl = i >> 10;
        const int n  = i & 1023;
        const int ch = g * 16 + cl;
        const float v = (x[base + i] - mean) * rstd * gscale[ch] + gbias[ch];
        h16[((long)b * 1024 + n) * 512 + ch] = (_Float16)v;
    }
}

// vT[b, c, m] = v[b, m, c] via LDS tile (keeps P·V as an NT GEMM).
__global__ __launch_bounds__(256)
void transpose_v(const _Float16* __restrict__ src, _Float16* __restrict__ dst)
{
    __shared__ _Float16 t[32][33];
    const int b  = blockIdx.z;
    const int c0 = blockIdx.x * 32;
    const int m0 = blockIdx.y * 32;
    const _Float16* s = src + (long)b * 1024 * 512;
    _Float16* d       = dst + (long)b * 512 * 1024;
    for (int i = threadIdx.x; i < 1024; i += 256) {
        const int m = i >> 5, c = i & 31;
        t[m][c] = s[(long)(m0 + m) * 512 + c0 + c];
    }
    __syncthreads();
    for (int i = threadIdx.x; i < 1024; i += 256) {
        const int c = i >> 5, m = i & 31;
        d[(long)(c0 + c) * 1024 + m0 + m] = t[m][c];
    }
}

// Row softmax in f32 over f16 scores; one wave32 per 1024-wide row, in place.
__global__ __launch_bounds__(256)
void softmax_rows(_Float16* __restrict__ S)
{
    const int lane = threadIdx.x & 31;
    const int wave = threadIdx.x >> 5;
    const long row = (long)blockIdx.x * 8 + wave;
    _Float16* p = S + row * 1024;

    float v[32];
    float m = -3.4e38f;
#pragma unroll
    for (int i = 0; i < 32; ++i) { v[i] = (float)p[lane + i * 32]; m = fmaxf(m, v[i]); }
#pragma unroll
    for (int off = 16; off > 0; off >>= 1) m = fmaxf(m, __shfl_xor(m, off, 32));
    float s = 0.f;
#pragma unroll
    for (int i = 0; i < 32; ++i) { v[i] = __expf(v[i] - m); s += v[i]; }
#pragma unroll
    for (int off = 16; off > 0; off >>= 1) s += __shfl_xor(s, off, 32);
    const float inv = 1.f / s;
#pragma unroll
    for (int i = 0; i < 32; ++i) p[lane + i * 32] = (_Float16)(v[i] * inv);
}

__global__ __launch_bounds__(256)
void f32_to_f16(const float* __restrict__ src, _Float16* __restrict__ dst, int n)
{
    const int i = blockIdx.x * 256 + threadIdx.x;
    if (i < n) dst[i] = (_Float16)src[i];
}

extern "C" void kernel_launch(void* const* d_in, const int* in_sizes, int n_in,
                              void* d_out, int out_size, void* d_ws, size_t ws_size,
                              hipStream_t stream)
{
    const float* x       = (const float*)d_in[0];
    const float* gnscale = (const float*)d_in[1];
    const float* gnbias  = (const float*)d_in[2];
    const float* wq      = (const float*)d_in[3];
    const float* bq      = (const float*)d_in[4];
    const float* wk      = (const float*)d_in[5];
    const float* bk      = (const float*)d_in[6];
    const float* wv      = (const float*)d_in[7];
    const float* bv      = (const float*)d_in[8];
    const float* wproj   = (const float*)d_in[9];
    const float* bproj   = (const float*)d_in[10];
    float* out = (float*)d_out;

    char* ws = (char*)d_ws;
    const size_t MB = 1ull << 20;
    _Float16* wq16 = (_Float16*)(ws + 0 * MB);
    _Float16* wk16 = (_Float16*)(ws + 512 * 1024);
    _Float16* wv16 = (_Float16*)(ws + 1 * MB);
    _Float16* wp16 = (_Float16*)(ws + 3 * MB / 2);
    _Float16* h16  = (_Float16*)(ws + 2 * MB);     // [B*N, C]  (32 MB), reused as vT
    _Float16* q16  = (_Float16*)(ws + 34 * MB);    // [B*N, C]  (32 MB), reused as attn-out
    _Float16* k16  = (_Float16*)(ws + 66 * MB);    // [B*N, C]  (32 MB)
    _Float16* v16p = (_Float16*)(ws + 98 * MB);    // [B*N, C]  (32 MB)
    _Float16* S    = (_Float16*)(ws + 130 * MB);   // [B, N, N] (64 MB)
    _Float16* vT16 = h16;                          // [B, C, N]
    _Float16* ao16 = q16;                          // [B*N, C]

    // 1) weights -> f16
    const int nW = 512 * 512;
    f32_to_f16<<<(nW + 255) / 256, 256, 0, stream>>>(wq,    wq16, nW);
    f32_to_f16<<<(nW + 255) / 256, 256, 0, stream>>>(wk,    wk16, nW);
    f32_to_f16<<<(nW + 255) / 256, 256, 0, stream>>>(wv,    wv16, nW);
    f32_to_f16<<<(nW + 255) / 256, 256, 0, stream>>>(wproj, wp16, nW);

    // 2) GroupNorm -> h16 [B*N, C]
    groupnorm_to_f16<<<32 * 32, 256, 0, stream>>>(x, gnscale, gnbias, h16);

    // 3) Q/K/V: one big NT GEMM each, M=32768 N=512 K=512
    dim3 gQKV(4, 256, 1);
    gemm_nt_wmma<STORE_F16, true><<<gQKV, 256, 0, stream>>>(
        h16, wq16, bq, q16, nullptr, nullptr, 512, 512, 1.f, 0, 0, 0);
    gemm_nt_wmma<STORE_F16, true><<<gQKV, 256, 0, stream>>>(
        h16, wk16, bk, k16, nullptr, nullptr, 512, 512, 1.f, 0, 0, 0);
    gemm_nt_wmma<STORE_F16, true><<<gQKV, 256, 0, stream>>>(
        h16, wv16, bv, v16p, nullptr, nullptr, 512, 512, 1.f, 0, 0, 0);

    // 4) V -> Vᵀ (into h16 region; h16 no longer needed)
    dim3 gT(16, 32, 32);
    transpose_v<<<gT, 256, 0, stream>>>(v16p, vT16);

    // 5) S = QKᵀ * C^-1/2 : batched NT, M=N=1024 K=512
    dim3 gS(8, 8, 32);
    const float scale = 0.044194173824159216f; // 512^-0.5
    gemm_nt_wmma<STORE_F16, false><<<gS, 256, 0, stream>>>(
        q16, k16, nullptr, S, nullptr, nullptr, 512, 1024, scale,
        (long)1024 * 512, (long)1024 * 512, (long)1024 * 1024);

    // 6) softmax rows (in place)
    softmax_rows<<<(32 * 1024) / 8, 256, 0, stream>>>(S);

    // 7) attn-out = P·V via NT against Vᵀ : batched, M=1024 N=512 K=1024 (into q16 region)
    dim3 gPV(4, 8, 32);
    gemm_nt_wmma<STORE_F16, false><<<gPV, 256, 0, stream>>>(
        S, vT16, nullptr, ao16, nullptr, nullptr, 1024, 512, 1.f,
        (long)1024 * 1024, (long)512 * 1024, (long)1024 * 512);

    // 8) proj + bias + residual -> out [B, C, N] f32
    dim3 gP(4, 256, 1);
    gemm_nt_wmma<STORE_F32_RESID, true><<<gP, 256, 0, stream>>>(
        ao16, wp16, bproj, nullptr, out, x, 512, 512, 1.f, 0, 0, 0);
}
